// MultiHeadSelfAttention_56504589746612
// MI455X (gfx1250) — matrix operand order repaired and checked by reading, not compiled, not run on hardware
//
#include <hip/hip_runtime.h>
#include <hip/hip_bf16.h>

// ---------------------------------------------------------------------------
// MI455X (gfx1250) multi-head self-attention, f16 WMMA with f32 accumulation.
// B=2, S=2048, D=1024, H=16, depth=64.
// ---------------------------------------------------------------------------

#define BATCH   2
#define SEQ     2048
#define DMODEL  1024
#define NHEADS  16
#define DEPTH   64
#define MROWS   (BATCH * SEQ)      // 4096

typedef _Float16 v16h __attribute__((ext_vector_type(16)));
typedef _Float16 v8h  __attribute__((ext_vector_type(8)));
typedef _Float16 v4h  __attribute__((ext_vector_type(4)));
typedef float    v8f  __attribute__((ext_vector_type(8)));

// ---------------------------------------------------------------------------
// WMMA fragment loaders (layouts per CDNA5 ISA 7.12.2, wave32).
// A frag 16x32 f16: lane l -> row m=l&15; halfs k = 8*(l>>4)+{0..7} and +{16..23}.
// B frag 32x16 f16: lane l -> col n=l&15; halfs k = 8*(l>>4)+{0..7} and +{16..23}.
// C/D frag 16x16 f32: VGPR r -> row r+8*(l>>4), col l&15.
// ---------------------------------------------------------------------------

__device__ __forceinline__ v16h load_a_frag(const _Float16* tile, int ld, int lane) {
    const int m  = lane & 15;
    const int ko = (lane >> 4) << 3;   // 0 or 8
    const _Float16* p = tile + (size_t)m * ld + ko;
    v8h lo = *(const v8h*)(p);
    v8h hi = *(const v8h*)(p + 16);
    return __builtin_shufflevector(lo, hi, 0,1,2,3,4,5,6,7,8,9,10,11,12,13,14,15);
}

__device__ __forceinline__ v16h load_b_frag(const _Float16* tile, int ld, int lane) {
    const int n  = lane & 15;
    const int ko = (lane >> 4) << 3;   // 0 or 8
    v16h r;
#pragma unroll
    for (int s = 0; s < 8; ++s) {
        r[s]     = tile[(size_t)(ko + s) * ld + n];
        r[s + 8] = tile[(size_t)(16 + ko + s) * ld + n];
    }
    return r;
}

__device__ __forceinline__ v8f wmma_f16(v16h a, v16h b, v8f c) {
    return __builtin_amdgcn_wmma_f32_16x16x32_f16(false, a, false, b,
                                                  (short)0, c, false, false);
}

// ---------------------------------------------------------------------------
// DPP16 cross-lane reductions within each 16-lane half (C-fragment rows).
// quad_perm ^1 (0xB1), quad_perm ^2 (0x4E), row_half_mirror ^7 (0x141),
// row_mirror ^15 (0x140). Pure VALU: no LDS pipeline, no dscnt waits.
// ---------------------------------------------------------------------------
template <int CTRL>
__device__ __forceinline__ float dpp_mov_f32(float x) {
    int xi = __float_as_int(x);
    int r  = __builtin_amdgcn_update_dpp(xi, xi, CTRL, 0xF, 0xF, false);
    return __int_as_float(r);
}

__device__ __forceinline__ float row16_max(float v) {
    v = fmaxf(v, dpp_mov_f32<0xB1>(v));
    v = fmaxf(v, dpp_mov_f32<0x4E>(v));
    v = fmaxf(v, dpp_mov_f32<0x141>(v));
    v = fmaxf(v, dpp_mov_f32<0x140>(v));
    return v;
}

__device__ __forceinline__ float row16_sum(float v) {
    v += dpp_mov_f32<0xB1>(v);
    v += dpp_mov_f32<0x4E>(v);
    v += dpp_mov_f32<0x141>(v);
    v += dpp_mov_f32<0x140>(v);
    return v;
}

// ---------------------------------------------------------------------------
// fp32 -> f16 conversion (float4 in, packed 8B out)
// ---------------------------------------------------------------------------
__global__ void cvt_f32_to_f16(const float* __restrict__ src,
                               _Float16* __restrict__ dst, int n4) {
    int i = blockIdx.x * blockDim.x + threadIdx.x;
    if (i < n4) {
        float4 v = ((const float4*)src)[i];
        v4h pk;
        pk[0] = (_Float16)v.x; pk[1] = (_Float16)v.y;
        pk[2] = (_Float16)v.z; pk[3] = (_Float16)v.w;
        *(v4h*)(dst + (size_t)i * 4) = pk;
    }
}

// ---------------------------------------------------------------------------
// Projection GEMM: C[4096,1024] = A[4096,1024] x B[1024,1024] + bias.
// Block = 256 threads (8 waves) arranged 4(M) x 2(N): block tile 128 x 128.
// Wave tile 32(M) x 64(N): 2 A frags x 4 B frags -> 8 WMMA per k-step.
// B k-slices (32x128, 8 KB) staged in LDS, double-buffered, one barrier per
// step; next slice's global load issued before compute to hide latency.
// MODE 0: f16 Q layout  [b][h][s][64]
// MODE 1: f16 Kt layout [b][h][64][s]   (packed 16B stores along s)
// MODE 2: f16 V layout  [b][h][s][64]
// MODE 3: f32 row-major [m][1024]
// ---------------------------------------------------------------------------
template <int MODE>
__global__ void __launch_bounds__(256)
gemm_proj(const _Float16* __restrict__ A, const _Float16* __restrict__ Bm,
          const float* __restrict__ bias, void* __restrict__ out) {
    __shared__ __align__(32) _Float16 Bls[2][32 * 128];    // 16 KB

    const int lane = threadIdx.x & 31;
    const int wave = threadIdx.x >> 5;
    const int wm   = wave >> 1;                 // 0..3
    const int wn   = wave & 1;                  // 0..1
    const int row0 = blockIdx.y * 128 + wm * 32;
    const int nblk = blockIdx.x * 128;
    const int n0   = nblk + wn * 64;

    const int tid  = threadIdx.x;
    const int brow = tid >> 3;                  // 0..31
    const int bcol = (tid & 7) << 4;            // 0..112

    v8f acc[2][4];
#pragma unroll
    for (int u = 0; u < 2; ++u)
#pragma unroll
        for (int t = 0; t < 4; ++t) acc[u][t] = (v8f)(0.0f);

    // prologue: fetch B k-slice 0 into registers
    v16h breg = *(const v16h*)(Bm + (size_t)brow * DMODEL + nblk + bcol);

    const int NSTEP = DMODEL / 32;              // 32
    for (int step = 0; step < NSTEP; ++step) {
        const int kk = step * 32;
        _Float16* bls = &Bls[step & 1][0];
        *(v16h*)&bls[brow * 128 + bcol] = breg;
        __syncthreads();                        // slice visible to all waves

        if (step + 1 < NSTEP)                   // overlap next slice with math
            breg = *(const v16h*)(Bm + (size_t)(kk + 32 + brow) * DMODEL + nblk + bcol);

        v16h af0 = load_a_frag(A + (size_t)row0 * DMODEL + kk, DMODEL, lane);
        v16h af1 = load_a_frag(A + (size_t)(row0 + 16) * DMODEL + kk, DMODEL, lane);
#pragma unroll
        for (int t = 0; t < 4; ++t) {
            v16h bf = load_b_frag(bls + wn * 64 + 16 * t, 128, lane);
            acc[0][t] = wmma_f16(af0, bf, acc[0][t]);
            acc[1][t] = wmma_f16(af1, bf, acc[1][t]);
        }
        // double buffer: next store targets the other slice; the barriers of
        // the next two iterations order reuse of this one.
    }

    const int hi = lane >> 4;
    const int lo = lane & 15;
#pragma unroll
    for (int u = 0; u < 2; ++u) {
        const int rbase = row0 + 16 * u + 8 * hi;       // rows rbase..rbase+7
        if constexpr (MODE == 1) {
            // Kt layout: for fixed lane, r=0..7 -> consecutive s. Pack 16B stores.
            const int b = rbase >> 11, s = rbase & (SEQ - 1);
            const size_t bb = (size_t)b * NHEADS;
            _Float16* o = (_Float16*)out;
#pragma unroll
            for (int t = 0; t < 4; ++t) {
                const int c = n0 + 16 * t + lo;          // 0..1023
                const int h = c >> 6, dd = c & 63;
                const float bv = bias[c];
                v8h pack;
#pragma unroll
                for (int r = 0; r < 8; ++r)
                    pack[r] = (_Float16)(acc[u][t][r] + bv);
                *(v8h*)&o[((bb + h) * DEPTH + dd) * SEQ + s] = pack;
            }
        } else {
#pragma unroll
            for (int r = 0; r < 8; ++r) {
                const int m = rbase + r;                 // global row
#pragma unroll
                for (int t = 0; t < 4; ++t) {
                    const int c   = n0 + 16 * t + lo;
                    const float v = acc[u][t][r] + bias[c];
                    if constexpr (MODE == 3) {
                        ((float*)out)[(size_t)m * DMODEL + c] = v;
                    } else {
                        const int b  = m >> 11, s = m & (SEQ - 1);
                        const int h  = c >> 6,  dd = c & 63;
                        _Float16* o  = (_Float16*)out;
                        o[(((size_t)(b * NHEADS + h)) * SEQ + s) * DEPTH + dd] = (_Float16)v;
                    }
                }
            }
        }
    }
}

// ---------------------------------------------------------------------------
// Flash attention: one block = one (b,h) x 128 q-rows; 8 waves x 16 rows.
// Q:[bh][s][64]  Kt:[bh][64][s]  V:[bh][s][64]  -> O merged f16 [b][s][1024]
// ---------------------------------------------------------------------------
__global__ void __launch_bounds__(256)
attn_flash(const _Float16* __restrict__ Q, const _Float16* __restrict__ Kt,
           const _Float16* __restrict__ V, _Float16* __restrict__ O) {
    __shared__ __align__(32) _Float16 Kls[DEPTH * 64];      // [dd][skv]  8 KB
    __shared__ __align__(32) _Float16 Vls[64 * DEPTH];      // [skv][dd]  8 KB
    __shared__ __align__(32) _Float16 Pls[8 * 16 * 64];     // per-wave   16 KB

    const int lane = threadIdx.x & 31;
    const int wave = threadIdx.x >> 5;
    const int hi   = lane >> 4;
    const int lo   = lane & 15;

    const int bh  = blockIdx.y;                 // 0..31
    const int b   = bh >> 4;
    const int h   = bh & 15;
    const int qw0 = blockIdx.x * 128 + wave * 16;

    const _Float16* Qbh  = Q  + (size_t)bh * SEQ * DEPTH;
    const _Float16* Ktbh = Kt + (size_t)bh * DEPTH * SEQ;
    const _Float16* Vbh  = V  + (size_t)bh * SEQ * DEPTH;

    // Q fragments for this wave's 16 rows, resident for the whole kernel.
    const v16h qf0 = load_a_frag(Qbh + (size_t)qw0 * DEPTH +  0, DEPTH, lane);
    const v16h qf1 = load_a_frag(Qbh + (size_t)qw0 * DEPTH + 32, DEPTH, lane);

    v8f o[4];
#pragma unroll
    for (int t = 0; t < 4; ++t) o[t] = (v8f)(0.0f);
    float m_i[8], l_i[8];
#pragma unroll
    for (int r = 0; r < 8; ++r) { m_i[r] = -3.0e38f; l_i[r] = 0.0f; }

    const float SCALE = 0.125f;                 // 1/sqrt(64)
    _Float16* pw = Pls + wave * (16 * 64);

    const int tid  = threadIdx.x;
    const int krow = tid >> 2;                  // 0..63
    const int c16  = (tid & 3) << 4;            // 0,16,32,48

    for (int s0 = 0; s0 < SEQ; s0 += 64) {
        __syncthreads();                        // previous tile fully consumed
        {   // cooperative K^T / V tile loads: 32 B per thread per tile
            const _Float16* kg = Ktbh + (size_t)krow * SEQ + s0 + c16;
            *(v8h*)&Kls[krow * 64 + c16]     = *(const v8h*)(kg);
            *(v8h*)&Kls[krow * 64 + c16 + 8] = *(const v8h*)(kg + 8);
            const _Float16* vg = Vbh + (size_t)(s0 + krow) * DEPTH + c16;
            *(v8h*)&Vls[krow * 64 + c16]     = *(const v8h*)(vg);
            *(v8h*)&Vls[krow * 64 + c16 + 8] = *(const v8h*)(vg + 8);
        }
        __syncthreads();

        // logits tile: 16(q) x 64(kv), accumulated over d = 64 (2 WMMA steps)
        v8f s[4];
#pragma unroll
        for (int t = 0; t < 4; ++t) {
            v8f z = (v8f)(0.0f);
            z = wmma_f16(qf0, load_b_frag(&Kls[ 0 * 64 + 16 * t], 64, lane), z);
            z = wmma_f16(qf1, load_b_frag(&Kls[32 * 64 + 16 * t], 64, lane), z);
            s[t] = z;
        }

        // online softmax; row stats via DPP16 (rows live in 16-lane halves)
#pragma unroll
        for (int r = 0; r < 8; ++r) {
            float mx = -3.0e38f;
#pragma unroll
            for (int t = 0; t < 4; ++t) { s[t][r] *= SCALE; mx = fmaxf(mx, s[t][r]); }
            mx = row16_max(mx);
            const float mnew = fmaxf(m_i[r], mx);
            const float corr = __expf(m_i[r] - mnew);
            float rsum = 0.0f;
#pragma unroll
            for (int t = 0; t < 4; ++t) {
                float p = __expf(s[t][r] - mnew);
                s[t][r] = p; rsum += p;
            }
            rsum = row16_sum(rsum);
            l_i[r] = l_i[r] * corr + rsum;
            m_i[r] = mnew;
#pragma unroll
            for (int t = 0; t < 4; ++t) o[t][r] *= corr;
        }

        // P: C-layout -> LDS -> A-fragment layout (wave-private region)
#pragma unroll
        for (int r = 0; r < 8; ++r)
#pragma unroll
            for (int t = 0; t < 4; ++t)
                pw[(r + 8 * hi) * 64 + 16 * t + lo] = (_Float16)s[t][r];
        asm volatile("s_wait_dscnt 0" ::: "memory");

        const v16h pf0 = load_a_frag(pw +  0, 64, lane);
        const v16h pf1 = load_a_frag(pw + 32, 64, lane);

        // O += P(16x64) x V(64x64)
#pragma unroll
        for (int t = 0; t < 4; ++t) {
            o[t] = wmma_f16(pf0, load_b_frag(&Vls[ 0 * 64 + 16 * t], 64, lane), o[t]);
            o[t] = wmma_f16(pf1, load_b_frag(&Vls[32 * 64 + 16 * t], 64, lane), o[t]);
        }
    }

    // normalize and store merged-head output [b][s][1024] as f16
#pragma unroll
    for (int r = 0; r < 8; ++r) {
        const float inv = 1.0f / l_i[r];
        const int sRow  = qw0 + r + 8 * hi;
#pragma unroll
        for (int t = 0; t < 4; ++t) {
            const int col = h * DEPTH + 16 * t + lo;
            O[((size_t)(b * SEQ + sRow)) * DMODEL + col] = (_Float16)(o[t][r] * inv);
        }
    }
}

// ---------------------------------------------------------------------------
// Host launcher
// ---------------------------------------------------------------------------
extern "C" void kernel_launch(void* const* d_in, const int* in_sizes, int n_in,
                              void* d_out, int out_size, void* d_ws, size_t ws_size,
                              hipStream_t stream) {
    const float* x  = (const float*)d_in[0];
    const float* wq = (const float*)d_in[1];
    const float* bq = (const float*)d_in[2];
    const float* wk = (const float*)d_in[3];
    const float* bk = (const float*)d_in[4];
    const float* wv = (const float*)d_in[5];
    const float* bv = (const float*)d_in[6];
    const float* wo = (const float*)d_in[7];
    const float* bo = (const float*)d_in[8];

    char* ws = (char*)d_ws;
    size_t off = 0;
    auto alloc_h = [&](size_t nhalfs) -> _Float16* {
        _Float16* p = (_Float16*)(ws + off);
        off = (off + nhalfs * sizeof(_Float16) + 255) & ~(size_t)255;
        return p;
    };
    const size_t XN = (size_t)MROWS * DMODEL;          // 4096*1024
    const size_t WN = (size_t)DMODEL * DMODEL;         // 1024*1024
    _Float16* Xh  = alloc_h(XN);
    _Float16* Wqh = alloc_h(WN);
    _Float16* Wkh = alloc_h(WN);
    _Float16* Wvh = alloc_h(WN);
    _Float16* Woh = alloc_h(WN);
    _Float16* Qb  = alloc_h(XN);                       // [b][h][s][64]
    _Float16* Ktb = alloc_h(XN);                       // [b][h][64][s]
    _Float16* Vb  = alloc_h(XN);                       // [b][h][s][64]
    _Float16* Oh  = alloc_h(XN);                       // [b][s][1024]

    {
        int n4 = (int)(XN / 4);
        cvt_f32_to_f16<<<(n4 + 255) / 256, 256, 0, stream>>>(x, Xh, n4);
        int w4 = (int)(WN / 4);
        cvt_f32_to_f16<<<(w4 + 255) / 256, 256, 0, stream>>>(wq, Wqh, w4);
        cvt_f32_to_f16<<<(w4 + 255) / 256, 256, 0, stream>>>(wk, Wkh, w4);
        cvt_f32_to_f16<<<(w4 + 255) / 256, 256, 0, stream>>>(wv, Wvh, w4);
        cvt_f32_to_f16<<<(w4 + 255) / 256, 256, 0, stream>>>(wo, Woh, w4);
    }

    dim3 gemmGrid(DMODEL / 128, MROWS / 128);          // (8, 32)
    gemm_proj<0><<<gemmGrid, 256, 0, stream>>>(Xh, Wqh, bq, (void*)Qb);
    gemm_proj<1><<<gemmGrid, 256, 0, stream>>>(Xh, Wkh, bk, (void*)Ktb);
    gemm_proj<2><<<gemmGrid, 256, 0, stream>>>(Xh, Wvh, bv, (void*)Vb);

    dim3 attnGrid(SEQ / 128, BATCH * NHEADS);          // (16, 32)
    attn_flash<<<attnGrid, 256, 0, stream>>>(Qb, Ktb, Vb, Oh);

    gemm_proj<3><<<gemmGrid, 256, 0, stream>>>(Oh, Woh, bo, d_out);
}
